// Llama_41661182771363
// MI455X (gfx1250) — matrix operand order repaired and checked by reading, not compiled, not run on hardware
//
#include <hip/hip_runtime.h>
#include <hip/hip_bf16.h>
#include <math.h>

#define DIMX    2048
#define NH      32
#define NKV     8
#define HD      64
#define NREP    4
#define KVDIM   512
#define HIDDENX 5632
#define VOCABX  32000
#define NLAYERS 4
#define BBATCH  2
#define SSEQ    1024
#define MT      (BBATCH*SSEQ)   // 2048 token rows
#define EPSX    1e-5f
#define NEGX    -1e30f

typedef __attribute__((ext_vector_type(16))) __bf16 bf16x16;
typedef __attribute__((ext_vector_type(8)))  float  v8f;

union FragU { bf16x16 v; uint4 q[2]; };

__device__ __forceinline__ unsigned short bf16bits(float x) {
    unsigned u = __float_as_uint(x);
    unsigned r = (u + 0x7FFFu + ((u >> 16) & 1u)) >> 16;   // RNE truncate
    return (unsigned short)r;
}
__device__ __forceinline__ unsigned pack2bf(float a, float b) {
    return (unsigned)bf16bits(a) | ((unsigned)bf16bits(b) << 16);
}

// ---------------------------------------------------------------- embedding
__global__ void embed_kernel(const int* __restrict__ tokens,
                             const float* __restrict__ embW,
                             float* __restrict__ h) {
    int row = blockIdx.x;
    int tok = tokens[row];
    const float* src = embW + (size_t)tok * DIMX;
    float* dst = h + (size_t)row * DIMX;
    for (int i = threadIdx.x; i < DIMX; i += blockDim.x) dst[i] = src[i];
}

// ---------------------------------------------------------------- rmsnorm
__global__ __launch_bounds__(256) void rmsnorm_kernel(const float* __restrict__ x,
                                                      const float* __restrict__ w,
                                                      float* __restrict__ y) {
    int row = blockIdx.x;
    const float* xr = x + (size_t)row * DIMX;
    float* yr = y + (size_t)row * DIMX;
    float v[8]; float ss = 0.f;
#pragma unroll
    for (int i = 0; i < 8; ++i) { v[i] = xr[threadIdx.x + 256 * i]; ss += v[i] * v[i]; }
    __shared__ float red[256];
    red[threadIdx.x] = ss; __syncthreads();
    for (int off = 128; off > 0; off >>= 1) {
        if (threadIdx.x < off) red[threadIdx.x] += red[threadIdx.x + off];
        __syncthreads();
    }
    float r = rsqrtf(red[0] / (float)DIMX + EPSX);
#pragma unroll
    for (int i = 0; i < 8; ++i) {
        int idx = threadIdx.x + 256 * i;
        yr[idx] = v[i] * r * w[idx];
    }
}

// ---------------------------------------------------------------- WMMA GEMM
// C[M,N] = A[M,K] * W[K,N] (+C if ADD). fp32 in, bf16 WMMA, fp32 out.
// Block tile 128x256, 8 waves (2M x 4N), 64x64/wave -> 16 WMMA per k-step.
// Double-buffered LDS, one barrier per k-step, packed b64 LDS staging.
// M%128==0, N%256==0, K%32==0 (all shapes here satisfy this).
#define BM 128
#define BN 256
#define BK 32
#define LP 40   // LDS row pitch in ushorts (80B: conflict-free b128 frag reads)

template <bool ADD>
__global__ __launch_bounds__(256) void gemm_kernel(const float* __restrict__ A,
                                                   const float* __restrict__ W,
                                                   float* __restrict__ C,
                                                   int M, int N, int K) {
    __shared__ unsigned short la[2][BM * LP];   // la[m][k]
    __shared__ unsigned short lb[2][BN * LP];   // lb[n][k] (transposed)
    int tid = threadIdx.x;
    int lane = tid & 31, wid = tid >> 5;
    int lo = lane & 15, hi = lane >> 4;
    int wm = wid >> 2, wn = wid & 3;            // 2 x 4 wave grid -> 64x64 per wave
    long m0 = (long)blockIdx.y * BM, n0 = (long)blockIdx.x * BN;

    float4 aReg[4];        // A staging: 4x float4 (row-major run of 4 k)
    float2 wReg[4][4];     // W staging: 4 units of (4k x 2n)

    auto loadTile = [&](int k0) {
#pragma unroll
        for (int it = 0; it < 4; ++it) {
            int flat4 = tid + 256 * it;
            int r = flat4 >> 3, c = (flat4 & 7) << 2;
            aReg[it] = *(const float4*)(A + (m0 + r) * (long)K + k0 + c);
        }
#pragma unroll
        for (int it = 0; it < 4; ++it) {
            int u = tid + 256 * it;
            int kk = (u >> 7) << 2, n = (u & 127) << 1;
#pragma unroll
            for (int e = 0; e < 4; ++e)
                wReg[it][e] = *(const float2*)(W + (long)(k0 + kk + e) * N + n0 + n);
        }
    };
    auto storeTile = [&](int buf) {
#pragma unroll
        for (int it = 0; it < 4; ++it) {
            int flat4 = tid + 256 * it;
            int r = flat4 >> 3, c = (flat4 & 7) << 2;
            uint2 t;
            t.x = pack2bf(aReg[it].x, aReg[it].y);
            t.y = pack2bf(aReg[it].z, aReg[it].w);
            *(uint2*)&la[buf][r * LP + c] = t;          // one b64 store
        }
#pragma unroll
        for (int it = 0; it < 4; ++it) {
            int u = tid + 256 * it;
            int kk = (u >> 7) << 2, n = (u & 127) << 1;
            uint2 t0, t1;
            t0.x = pack2bf(wReg[it][0].x, wReg[it][1].x);
            t0.y = pack2bf(wReg[it][2].x, wReg[it][3].x);
            t1.x = pack2bf(wReg[it][0].y, wReg[it][1].y);
            t1.y = pack2bf(wReg[it][2].y, wReg[it][3].y);
            *(uint2*)&lb[buf][(n + 0) * LP + kk] = t0;  // transposed b64 stores
            *(uint2*)&lb[buf][(n + 1) * LP + kk] = t1;
        }
    };

    v8f acc[4][4] = {};
    loadTile(0);
    storeTile(0);
    for (int k0 = 0; k0 < K; k0 += BK) {
        int cur = (k0 / BK) & 1;
        __syncthreads();                         // buf[cur] ready; buf[cur^1] free
        bool more = (k0 + BK) < K;
        if (more) loadTile(k0 + BK);             // issue next-tile global loads early

        FragU af[4];
#pragma unroll
        for (int i = 0; i < 4; ++i) {            // A frags: lane=row, K split {8hi},{16+8hi}
            int mp = wm * 64 + i * 16 + lo;
            const char* base = (const char*)&la[cur][0] + mp * (LP * 2);
            af[i].q[0] = *(const uint4*)(base + 16 * hi);
            af[i].q[1] = *(const uint4*)(base + 32 + 16 * hi);
        }
#pragma unroll
        for (int j = 0; j < 4; ++j) {            // stream B frags, reuse across 4 WMMAs
            FragU bfr;
            int np = wn * 64 + j * 16 + lo;
            const char* base = (const char*)&lb[cur][0] + np * (LP * 2);
            bfr.q[0] = *(const uint4*)(base + 16 * hi);
            bfr.q[1] = *(const uint4*)(base + 32 + 16 * hi);
#pragma unroll
            for (int i = 0; i < 4; ++i)
                acc[i][j] = __builtin_amdgcn_wmma_f32_16x16x32_bf16(
                    false, af[i].v, false, bfr.v, (short)0, acc[i][j], false, false);
        }
        if (more) storeTile(cur ^ 1);            // convert+stage after compute
    }

    // branch-free epilogue, incremental addressing
#pragma unroll
    for (int i = 0; i < 4; ++i)
#pragma unroll
        for (int j = 0; j < 4; ++j) {
            long row = m0 + wm * 64 + i * 16 + 8 * hi;
            long col = n0 + wn * 64 + j * 16 + lo;
            float* p = C + row * (long)N + col;
            if (ADD) {
                float old[8];
#pragma unroll
                for (int jj = 0; jj < 8; ++jj) old[jj] = p[(long)jj * N];
#pragma unroll
                for (int jj = 0; jj < 8; ++jj) p[(long)jj * N] = old[jj] + acc[i][j][jj];
            } else {
#pragma unroll
                for (int jj = 0; jj < 8; ++jj) p[(long)jj * N] = acc[i][j][jj];
            }
        }
}

// ---------------------------------------------------------------- RoPE (fp32 -> bf16)
__global__ void rope_kernel(const float* __restrict__ x,
                            const float* __restrict__ fc, const float* __restrict__ fs,
                            unsigned short* __restrict__ y, int nheads, int total) {
    int gid = blockIdx.x * blockDim.x + threadIdx.x;
    if (gid >= total) return;
    int d2 = gid & 31;
    int hh = (gid >> 5) % nheads;
    int bs = gid / (32 * nheads);
    int s  = bs % SSEQ;
    long base = ((long)bs * nheads + hh) * HD + 2 * d2;
    float a = x[base], b = x[base + 1];
    float c = fc[s * 32 + d2], sn = fs[s * 32 + d2];
    y[base]     = bf16bits(a * c - b * sn);
    y[base + 1] = bf16bits(a * sn + b * c);
}

__global__ void f2bf_kernel(const float* __restrict__ x, unsigned short* __restrict__ y, int total) {
    int gid = blockIdx.x * blockDim.x + threadIdx.x;
    if (gid < total) y[gid] = bf16bits(x[gid]);
}

__global__ void silu_mul_kernel(float* __restrict__ h1, const float* __restrict__ h3, int total) {
    int gid = blockIdx.x * blockDim.x + threadIdx.x;
    if (gid < total) {
        float a = h1[gid];
        h1[gid] = (a / (1.f + __expf(-a))) * h3[gid];
    }
}

// ---------------------------------------------------------------- flash attention (WMMA)
// 1 wave per (b, head, 16-query tile). bf16 q/k/v, fp32 out. Causal.
#define PLP 40
__global__ __launch_bounds__(32) void attn_kernel(const unsigned short* __restrict__ qmat,
                                                  const unsigned short* __restrict__ kmat,
                                                  const unsigned short* __restrict__ vmat,
                                                  float* __restrict__ out, float scale) {
    __shared__ unsigned short pl[16 * PLP];   // P   [16 q][32 key]
    __shared__ unsigned short vt[64 * PLP];   // V^T [64 d][32 key]
    int lane = threadIdx.x;
    int lo = lane & 15, hi = lane >> 4;
    int qt = blockIdx.x & (SSEQ / 16 - 1);            // 64 q tiles
    int hq = (blockIdx.x >> 6) & (NH - 1);
    int b  = blockIdx.x >> 11;
    int kvh = hq / NREP;
    int qbase = qt * 16;

    // Q A-fragments for the two 32-wide d steps (load once)
    FragU aq[2];
    const unsigned short* qrow = qmat + (((long)b * SSEQ + qbase + lo) * NH + hq) * HD;
#pragma unroll
    for (int s = 0; s < 2; ++s) {
        aq[s].q[0] = *(const uint4*)(qrow + 32 * s + 8 * hi);
        aq[s].q[1] = *(const uint4*)(qrow + 32 * s + 16 + 8 * hi);
    }

    float mrow[8], lrow[8];
#pragma unroll
    for (int j = 0; j < 8; ++j) { mrow[j] = -3.0e38f; lrow[j] = 0.f; }
    v8f o[4] = {};

    int kq = lane & 7;            // key group: keys 4kq..4kq+3
    int db = (lane >> 3) << 4;    // d range:  db..db+15

    for (int kb0 = 0; kb0 < qbase + 16; kb0 += 32) {
        // stage V tile transposed into LDS: vt[d][key], packed b64 stores
        union RowU { uint4 q[2]; unsigned short s[16]; } rr[4];
#pragma unroll
        for (int e = 0; e < 4; ++e) {
            const unsigned short* vr =
                vmat + (((long)b * SSEQ + kb0 + 4 * kq + e) * NKV + kvh) * HD + db;
            rr[e].q[0] = *(const uint4*)(vr);
            rr[e].q[1] = *(const uint4*)(vr + 8);
        }
#pragma unroll
        for (int i = 0; i < 16; ++i) {
            uint2 t;
            t.x = (unsigned)rr[0].s[i] | ((unsigned)rr[1].s[i] << 16);
            t.y = (unsigned)rr[2].s[i] | ((unsigned)rr[3].s[i] << 16);
            *(uint2*)&vt[(db + i) * PLP + 4 * kq] = t;
        }

        // scores S[16q x 32key] = Q * K^T  (K rows already in B-frag layout)
        v8f sc[2] = {};
#pragma unroll
        for (int t = 0; t < 2; ++t) {
            const unsigned short* krow =
                kmat + (((long)b * SSEQ + kb0 + 16 * t + lo) * NKV + kvh) * HD;
#pragma unroll
            for (int s = 0; s < 2; ++s) {
                FragU bk;
                bk.q[0] = *(const uint4*)(krow + 32 * s + 8 * hi);
                bk.q[1] = *(const uint4*)(krow + 32 * s + 16 + 8 * hi);
                sc[t] = __builtin_amdgcn_wmma_f32_16x16x32_bf16(
                    false, aq[s].v, false, bk.v, (short)0, sc[t], false, false);
            }
        }

        // scale + causal mask + online softmax (row = jj + 8*hi in C layout)
#pragma unroll
        for (int j = 0; j < 8; ++j) {
            int row = qbase + j + 8 * hi;
            float s0 = sc[0][j] * scale; if (kb0 + lo > row)      s0 = NEGX;
            float s1 = sc[1][j] * scale; if (kb0 + 16 + lo > row) s1 = NEGX;
            float tm = fmaxf(s0, s1);
            for (int m = 1; m < 16; m <<= 1) tm = fmaxf(tm, __shfl_xor(tm, m, 32));
            float nm = fmaxf(mrow[j], tm);
            float alpha = __expf(mrow[j] - nm);
            mrow[j] = nm;
            float p0 = __expf(s0 - nm), p1 = __expf(s1 - nm);
            float rs = p0 + p1;
            for (int m = 1; m < 16; m <<= 1) rs += __shfl_xor(rs, m, 32);
            lrow[j] = lrow[j] * alpha + rs;
#pragma unroll
            for (int t = 0; t < 4; ++t) o[t][j] *= alpha;
            pl[(j + 8 * hi) * PLP + lo]      = bf16bits(p0);
            pl[(j + 8 * hi) * PLP + 16 + lo] = bf16bits(p1);
        }
        __syncthreads();

        // P as A-fragment, V^T columns as B-fragments; O += P * V
        FragU pf;
        {
            const char* base = (const char*)pl + lo * (PLP * 2);
            pf.q[0] = *(const uint4*)(base + 16 * hi);
            pf.q[1] = *(const uint4*)(base + 32 + 16 * hi);
        }
#pragma unroll
        for (int t = 0; t < 4; ++t) {
            FragU vf;
            const char* base = (const char*)vt + (16 * t + lo) * (PLP * 2);
            vf.q[0] = *(const uint4*)(base + 16 * hi);
            vf.q[1] = *(const uint4*)(base + 32 + 16 * hi);
            o[t] = __builtin_amdgcn_wmma_f32_16x16x32_bf16(
                false, pf.v, false, vf.v, (short)0, o[t], false, false);
        }
        __syncthreads();
    }

#pragma unroll
    for (int t = 0; t < 4; ++t)
#pragma unroll
        for (int j = 0; j < 8; ++j) {
            long row = (long)b * SSEQ + qbase + j + 8 * hi;
            out[row * DIMX + hq * HD + 16 * t + lo] = o[t][j] / lrow[j];
        }
}

// ---------------------------------------------------------------- launcher
extern "C" void kernel_launch(void* const* d_in, const int* in_sizes, int n_in,
                              void* d_out, int out_size, void* d_ws, size_t ws_size,
                              hipStream_t stream) {
    (void)in_sizes; (void)n_in; (void)out_size; (void)ws_size;
    const int*   tokens = (const int*)  d_in[0];
    const float* fc     = (const float*)d_in[2];
    const float* fs     = (const float*)d_in[3];
    const float* embW   = (const float*)d_in[5];
    const float* wq     = (const float*)d_in[6];
    const float* wk     = (const float*)d_in[7];
    const float* wv     = (const float*)d_in[8];
    const float* wo     = (const float*)d_in[9];
    const float* w1     = (const float*)d_in[10];
    const float* w2     = (const float*)d_in[11];
    const float* w3     = (const float*)d_in[12];
    const float* anw    = (const float*)d_in[13];
    const float* fnw    = (const float*)d_in[14];
    const float* nw     = (const float*)d_in[15];
    const float* outW   = (const float*)d_in[16];

    char* p = (char*)d_ws;
    float* h   = (float*)p;          p += (size_t)MT * DIMX   * 4;
    float* xn  = (float*)p;          p += (size_t)MT * DIMX   * 4;
    float* qf  = (float*)p;          p += (size_t)MT * DIMX   * 4;
    float* kf  = (float*)p;          p += (size_t)MT * KVDIM  * 4;
    float* vf  = (float*)p;          p += (size_t)MT * KVDIM  * 4;
    unsigned short* qbf = (unsigned short*)p; p += (size_t)MT * DIMX  * 2;
    unsigned short* kbf = (unsigned short*)p; p += (size_t)MT * KVDIM * 2;
    unsigned short* vbf = (unsigned short*)p; p += (size_t)MT * KVDIM * 2;
    float* att = (float*)p;          p += (size_t)MT * DIMX   * 4;
    float* h1  = (float*)p;          p += (size_t)MT * HIDDENX * 4;
    float* h3  = (float*)p;          p += (size_t)MT * HIDDENX * 4;

    const float scale = 0.125f;  // 1/sqrt(HEAD_DIM)

    embed_kernel<<<MT, 256, 0, stream>>>(tokens, embW, h);

    for (int l = 0; l < NLAYERS; ++l) {
        const float* wq_ = wq + (size_t)l * DIMX    * DIMX;
        const float* wk_ = wk + (size_t)l * DIMX    * KVDIM;
        const float* wv_ = wv + (size_t)l * DIMX    * KVDIM;
        const float* wo_ = wo + (size_t)l * DIMX    * DIMX;
        const float* w1_ = w1 + (size_t)l * DIMX    * HIDDENX;
        const float* w2_ = w2 + (size_t)l * HIDDENX * DIMX;
        const float* w3_ = w3 + (size_t)l * DIMX    * HIDDENX;

        rmsnorm_kernel<<<MT, 256, 0, stream>>>(h, anw + (size_t)l * DIMX, xn);
        gemm_kernel<false><<<dim3(DIMX / BN,  MT / BM), 256, 0, stream>>>(xn, wq_, qf, MT, DIMX,  DIMX);
        gemm_kernel<false><<<dim3(KVDIM / BN, MT / BM), 256, 0, stream>>>(xn, wk_, kf, MT, KVDIM, DIMX);
        gemm_kernel<false><<<dim3(KVDIM / BN, MT / BM), 256, 0, stream>>>(xn, wv_, vf, MT, KVDIM, DIMX);

        int qtot = MT * NH  * 32;
        int ktot = MT * NKV * 32;
        int vtot = MT * KVDIM;
        rope_kernel<<<(qtot + 255) / 256, 256, 0, stream>>>(qf, fc, fs, qbf, NH,  qtot);
        rope_kernel<<<(ktot + 255) / 256, 256, 0, stream>>>(kf, fc, fs, kbf, NKV, ktot);
        f2bf_kernel<<<(vtot + 255) / 256, 256, 0, stream>>>(vf, vbf, vtot);

        attn_kernel<<<BBATCH * NH * (SSEQ / 16), 32, 0, stream>>>(qbf, kbf, vbf, att, scale);

        gemm_kernel<true><<<dim3(DIMX / BN, MT / BM), 256, 0, stream>>>(att, wo_, h, MT, DIMX, DIMX);

        rmsnorm_kernel<<<MT, 256, 0, stream>>>(h, fnw + (size_t)l * DIMX, xn);
        gemm_kernel<false><<<dim3(HIDDENX / BN, MT / BM), 256, 0, stream>>>(xn, w1_, h1, MT, HIDDENX, DIMX);
        gemm_kernel<false><<<dim3(HIDDENX / BN, MT / BM), 256, 0, stream>>>(xn, w3_, h3, MT, HIDDENX, DIMX);
        int ftot = MT * HIDDENX;
        silu_mul_kernel<<<(ftot + 255) / 256, 256, 0, stream>>>(h1, h3, ftot);
        gemm_kernel<true><<<dim3(DIMX / BN, MT / BM), 256, 0, stream>>>(h1, w2_, h, MT, DIMX, HIDDENX);
    }

    rmsnorm_kernel<<<MT, 256, 0, stream>>>(h, nw, xn);
    gemm_kernel<false><<<dim3(VOCABX / BN, MT / BM), 256, 0, stream>>>(xn, outW, (float*)d_out, MT, VOCABX, DIMX);
}
